// PathLoss_64613488001157
// MI455X (gfx1250) — matrix operand order repaired and checked by reading, not compiled, not run on hardware
//
#include <hip/hip_runtime.h>
#include <hip/hip_bf16.h>

// ---------------------------------------------------------------------------
// PathLoss (batched cosine-sim CE) for MI455X / gfx1250.
//
// Roofline: GEMM = 2*B*M*N*d = 34.4 GFLOP; f32 inputs = 192 MB -> ~8 us HBM
// floor at 23.3 TB/s. Strategy: read f32 once (normalize -> bf16 workspace),
// never materialize the 64 MB logits (fused online softmax), run the GEMM on
// v_wmma_f32_16x16x32_bf16. Each wave owns TWO 16x16 output tiles that share
// one A fragment (2x data reuse, 2 independent WMMA chains). A tile is staged
// into LDS with GLOBAL_LOAD_ASYNC_TO_LDS_B128 (ASYNCcnt path, no VGPR bounce).
// ---------------------------------------------------------------------------

typedef __bf16 bf16_t;
typedef bf16_t v16bf __attribute__((ext_vector_type(16)));
typedef bf16_t v8bf  __attribute__((ext_vector_type(8)));
typedef float  v8f   __attribute__((ext_vector_type(8)));

#define B_ 32
#define M_ 512
#define N_ 1024
#define D_ 1024

__device__ __forceinline__ unsigned short f32_to_bf16_bits(float x) {
  unsigned u = __float_as_uint(x);
  u += 0x7FFFu + ((u >> 16) & 1u);   // round-to-nearest-even
  return (unsigned short)(u >> 16);
}

// ---------------------------------------------------------------------------
// Kernel 0: zero the scalar accumulator (d_out is poisoned by the harness).
// ---------------------------------------------------------------------------
__global__ void zero_out_kernel(float* out) { out[0] = 0.0f; }

// ---------------------------------------------------------------------------
// Kernel 1: row-wise L2 normalize (matches F.normalize eps=1e-12) + f32->bf16.
// One 256-thread block per row of 1024 floats (im rows then s rows).
// ---------------------------------------------------------------------------
__global__ void normalize_bf16_kernel(const float* __restrict__ im,
                                      const float* __restrict__ s,
                                      unsigned short* __restrict__ im_bf,
                                      unsigned short* __restrict__ s_bf) {
  const int row = blockIdx.x;
  const float* src;
  unsigned short* dst;
  if (row < B_ * M_) {
    src = im + (size_t)row * D_;
    dst = im_bf + (size_t)row * D_;
  } else {
    const size_t r = (size_t)(row - B_ * M_);
    src = s + r * D_;
    dst = s_bf + r * D_;
  }
  const int t = threadIdx.x;
  float4 v = ((const float4*)src)[t];          // 256 threads * 4 = 1024
  float ss = v.x * v.x + v.y * v.y + v.z * v.z + v.w * v.w;
  #pragma unroll
  for (int m = 16; m > 0; m >>= 1) ss += __shfl_xor(ss, m, 32);  // wave32
  __shared__ float red[8];
  if ((t & 31) == 0) red[t >> 5] = ss;
  __syncthreads();
  if (t == 0) {
    float a = 0.f;
    #pragma unroll
    for (int i = 0; i < 8; ++i) a += red[i];
    red[0] = 1.0f / fmaxf(sqrtf(a), 1e-12f);
  }
  __syncthreads();
  const float inv = red[0];
  unsigned long long p =
      (unsigned long long)f32_to_bf16_bits(v.x * inv) |
      ((unsigned long long)f32_to_bf16_bits(v.y * inv) << 16) |
      ((unsigned long long)f32_to_bf16_bits(v.z * inv) << 32) |
      ((unsigned long long)f32_to_bf16_bits(v.w * inv) << 48);
  ((unsigned long long*)dst)[t] = p;
}

// ---------------------------------------------------------------------------
// Kernel 2: fused 16xN GEMM tile + masked online log-softmax + CE gather.
// Block = 256 threads (8 waves) handles (batch b, 16-row tile m0).
// 4 passes of 256 columns; wave w computes cols [w*32, w*32+32) as two 16x16
// WMMA tiles sharing the A fragment. K = 1024 -> 32 wmma pairs per pass.
// ---------------------------------------------------------------------------
__global__ void __launch_bounds__(256)
path_loss_kernel(const unsigned short* __restrict__ im_bf,
                 const unsigned short* __restrict__ s_bf,
                 const long long* __restrict__ paths,
                 const long long* __restrict__ len_a,
                 const long long* __restrict__ len_b,
                 float* __restrict__ out) {
  const int b    = blockIdx.x >> 5;   // M_/16 = 32 row tiles per batch
  const int mt   = blockIdx.x & 31;
  const int m0   = mt * 16;
  const int tid  = threadIdx.x;
  const int lane = tid & 31;
  const int wave = tid >> 5;

  // A tile padded to 1032 bf16/row (2064 B = 516 dwords == 4 banks mod 64).
  __shared__ unsigned short Atile[16 * 1032];
  __shared__ float scores[16 * 260];            // 16 rows x 256 cols (+pad)
  __shared__ float rowMax[16], rowSum[16], rowT[16];

  // Async A-tile stage: 16 rows x 1024 bf16 = 2048 chunks of 16 B, moved with
  // GLOBAL_LOAD_ASYNC_TO_LDS_B128 (tracked by ASYNCcnt; LDS offset is the low
  // 32 bits of the generic pointer per the LDS aperture mapping).
  for (int c = tid; c < 2048; c += 256) {
    const int r  = c >> 7;                      // 128 chunks per row
    const int cb = (c & 127) << 3;              // bf16 column base
    unsigned long long ga = (unsigned long long)(uintptr_t)(
        im_bf + ((size_t)(b * M_ + m0 + r) * D_ + cb));
    unsigned ldsOff = (unsigned)(uintptr_t)(&Atile[r * 1032 + cb]);
    asm volatile("global_load_async_to_lds_b128 %0, %1, off"
                 :: "v"(ldsOff), "v"(ga) : "memory");
  }
  if (tid < 16) { rowMax[tid] = -1e30f; rowSum[tid] = 0.0f; rowT[tid] = 0.0f; }
  asm volatile("s_wait_asynccnt 0x0" ::: "memory");
  __syncthreads();

  const int lenb = (int)len_b[b];
  const int rowA = lane & 15;     // A row this lane supplies
  const int hsel = lane >> 4;     // half-wave select
  const int kh   = hsel << 4;     // A layout: lanes16-31 hold K+16 (contiguous, matches B)

  for (int pass = 0; pass < 4; ++pass) {
    const int cbase = pass * 256 + wave * 32 + (lane & 15);
    // B fragment: 32x16 bf16 layout -> lane holds 16 contiguous K of its
    // column; lanes 16-31 shifted by K+16. Column of B == row of s_hat.
    const unsigned short* bcol0 =
        s_bf + (size_t)(b * N_ + cbase) * D_ + (hsel << 4);
    const unsigned short* bcol1 = bcol0 + (size_t)16 * D_;

    v8f acc0 = {}, acc1 = {};
    #pragma unroll 4
    for (int k0 = 0; k0 < D_; k0 += 32) {
      const int abase = rowA * 1032 + k0 + kh;
      v8bf alo = *(const v8bf*)(&Atile[abase]);
      v8bf ahi = *(const v8bf*)(&Atile[abase + 8]);
      v16bf afrag = __builtin_shufflevector(
          alo, ahi, 0, 1, 2, 3, 4, 5, 6, 7, 8, 9, 10, 11, 12, 13, 14, 15);
      v16bf bf0 = *(const v16bf*)(bcol0 + k0);
      v16bf bf1 = *(const v16bf*)(bcol1 + k0);
      // Branchless prefetch four iterations ahead, wrapped within the row.
      __builtin_prefetch(bcol0 + ((k0 + 128) & (D_ - 1)), 0, 1);
      __builtin_prefetch(bcol1 + ((k0 + 128) & (D_ - 1)), 0, 1);
      acc0 = __builtin_amdgcn_wmma_f32_16x16x32_bf16(
          false, afrag, false, bf0, (short)0, acc0, false, false);
      acc1 = __builtin_amdgcn_wmma_f32_16x16x32_bf16(
          false, afrag, false, bf1, (short)0, acc1, false, false);
    }

    // D layout: VGPR j, lanes0-15 -> row j, lanes16-31 -> row j+8; col=lane%16.
    #pragma unroll
    for (int j = 0; j < 8; ++j) {
      const int r = (hsel << 3) + j;
      const int c = wave * 32 + (lane & 15);
      scores[r * 260 + c]      = (1.0f + acc0[j]) * 0.5f;   // D = (1+cos)/2
      scores[r * 260 + c + 16] = (1.0f + acc1[j]) * 0.5f;
    }
    __syncthreads();

    // Online softmax merge: 16 threads per row, 16 columns each.
    {
      const int r    = tid >> 4;
      const int c0   = tid & 15;
      const int rowg = m0 + r;
      const int pt   = (int)paths[(size_t)b * M_ + rowg];
      float lm = -1e30f, ls = 0.0f;
      #pragma unroll
      for (int i = 0; i < 16; ++i) {
        const int c  = c0 + (i << 4);
        const int gc = pass * 256 + c;
        if (gc < lenb) {                       // column mask (len_b)
          const float x = scores[r * 260 + c];
          if (gc == pt) rowT[r] = x;           // target logit (always valid)
          if (x > lm) { ls = ls * __expf(lm - x) + 1.0f; lm = x; }
          else        { ls += __expf(x - lm); }
        }
      }
      #pragma unroll
      for (int mdel = 8; mdel > 0; mdel >>= 1) {  // width-16 segment reduce
        const float om = __shfl_xor(lm, mdel, 16);
        const float os = __shfl_xor(ls, mdel, 16);
        const float nm = fmaxf(lm, om);
        ls = ls * __expf(lm - nm) + os * __expf(om - nm);
        lm = nm;
      }
      if (c0 == 0) {                            // one writer per row
        const float om = rowMax[r], os = rowSum[r];
        const float nm = fmaxf(lm, om);
        rowSum[r] = os * __expf(om - nm) + ls * __expf(lm - nm);
        rowMax[r] = nm;
      }
    }
    __syncthreads();
  }

  // CE per row, masked by len_a, weighted mean -> scalar.
  if (tid < 16) {
    const int rowg = m0 + tid;
    const long long la = len_a[b];
    if (rowg < (int)la) {
      const float ce = __logf(rowSum[tid]) + rowMax[tid] - rowT[tid];
      atomicAdd(out, ce / ((float)la * (float)B_));
    }
  }
}

// ---------------------------------------------------------------------------
// Launcher
// ---------------------------------------------------------------------------
extern "C" void kernel_launch(void* const* d_in, const int* in_sizes, int n_in,
                              void* d_out, int out_size, void* d_ws, size_t ws_size,
                              hipStream_t stream) {
  const float*     im     = (const float*)d_in[0];
  const float*     s      = (const float*)d_in[1];
  const long long* paths  = (const long long*)d_in[2];
  const long long* len_a  = (const long long*)d_in[3];
  const long long* len_b  = (const long long*)d_in[4];
  float*           out    = (float*)d_out;

  unsigned short* im_bf = (unsigned short*)d_ws;                       // 32 MB
  unsigned short* s_bf  = im_bf + (size_t)B_ * M_ * D_;                // 64 MB

  zero_out_kernel<<<1, 1, 0, stream>>>(out);
  normalize_bf16_kernel<<<B_ * (M_ + N_), 256, 0, stream>>>(im, s, im_bf, s_bf);
  path_loss_kernel<<<B_ * (M_ / 16), 256, 0, stream>>>(im_bf, s_bf, paths,
                                                       len_a, len_b, out);
}
